// Qwen3Attention_4148938408203
// MI455X (gfx1250) — matrix operand order repaired and checked by reading, not compiled, not run on hardware
//
#include <hip/hip_runtime.h>
#include <hip/hip_bf16.h>
#include <math.h>

// ---------------------------------------------------------------------------
// Qwen3 attention layer for gfx1250 (MI455X).
// bf16 WMMA pipeline + Tensor Data Mover (TDM) LDS staging.
// ---------------------------------------------------------------------------

typedef __bf16 bf16;
typedef __bf16 v16bf __attribute__((ext_vector_type(16)));
typedef float  v8f   __attribute__((ext_vector_type(8)));
typedef unsigned int v4u __attribute__((ext_vector_type(4)));
typedef int v8i __attribute__((ext_vector_type(8)));
typedef int v4i __attribute__((ext_vector_type(4)));

constexpr int NQ     = 32;     // query heads
constexpr int NKV    = 8;      // kv heads
constexpr int HD     = 128;    // head dim
constexpr int HID    = 4096;   // hidden
constexpr int BATCH  = 2;
constexpr int SEQ    = 1024;
constexpr int T      = BATCH * SEQ;   // 2048 tokens
constexpr int KVROWS = 16384;         // kv_buffer rows

#if defined(__gfx1250__) && __has_builtin(__builtin_amdgcn_tensor_load_to_lds)
#define USE_TDM 1
#else
#define USE_TDM 0
#endif

// generic-pointer low 32 bits == LDS byte offset (flat aperture scheme)
static __device__ __forceinline__ unsigned lds_off32(const void* p) {
  return (unsigned)(unsigned long long)p;
}

#if USE_TDM
// 2-D tile DMA: global (row-major, row_stride elems of 2 bytes) -> LDS packed.
// D# per CDNA5 ISA ch.8: group0 {count=1, lds_addr, global_addr, type=2},
// group1 {data_size=1(2B), tensor dims (huge: tile always in-bounds),
//         tile_dim0 = tile_w, tile_dim1 = tile_h, tensor_dim0_stride}.
// 6-arg builtin form (clang-23 / therock-10.0 lane).
static __device__ __forceinline__ void tdm_load_2d(unsigned lds_byte_off,
                                                   const void* gsrc,
                                                   unsigned tile_w, unsigned tile_h,
                                                   unsigned long long row_stride) {
  unsigned long long ga = (unsigned long long)gsrc;
  v4u g0;
  g0[0] = 1u;                                            // count=1 user D#
  g0[1] = lds_byte_off;                                  // lds_addr
  g0[2] = (unsigned)ga;                                  // global_addr[31:0]
  g0[3] = (unsigned)((ga >> 32) & 0x01FFFFFFull) | (2u << 30); // [56:32]|type=2
  v8i g1;
  g1[0] = (int)(1u << 16);                               // data_size=2B, mask=0
  g1[1] = 0;                                             // tensor_dim0[15:0]=0
  g1[2] = (int)0x4000;                                   // tensor_dim0=2^30
  g1[3] = (int)(0x4000u | (tile_w << 16));               // tensor_dim1=2^30|tile_dim0
  g1[4] = (int)tile_h;                                   // tile_dim1 (tile_dim2=0)
  g1[5] = (int)(unsigned)row_stride;                     // dim0_stride[31:0]
  g1[6] = (int)(unsigned)(row_stride >> 32);             // dim0_stride[47:32]
  g1[7] = 0;
  v4i gz4 = {0, 0, 0, 0};
  v8i gz8 = {0, 0, 0, 0, 0, 0, 0, 0};
  __builtin_amdgcn_tensor_load_to_lds(g0, g1, gz4, gz4, gz8, 0);
}
#endif

// ---------------- f32 -> bf16 (plain) ----------------
__global__ __launch_bounds__(256) void k_cvt(const float* __restrict__ in,
                                             bf16* __restrict__ out, int n) {
  int i = blockIdx.x * 256 + threadIdx.x;
  if (i < n) out[i] = (bf16)in[i];
}

// ---------------- f32 [R,C] -> bf16 transposed [C,R] ----------------
__global__ __launch_bounds__(256) void k_cvt_t(const float* __restrict__ in,
                                               bf16* __restrict__ out,
                                               int R, int C) {
  int i = blockIdx.x * 256 + threadIdx.x;
  if (i < R * C) {
    int r = i / C, c = i % C;
    out[(size_t)c * R + r] = (bf16)in[i];
  }
}

// ---------------- bf16 GEMM: C[M,N] = A[M,K] * B[K,N], B given as BT[N,K] ---
// grid = (N/128, M/128), 256 thr (8 waves); wave (wm,wn) -> 64x32 sub-tile.
// Double-buffered LDS tiles, staged by TDM (wave 0 issues DMA).
__global__ __launch_bounds__(256) void k_gemm(const bf16* __restrict__ A,
                                              const bf16* __restrict__ BT,
                                              float* __restrict__ C,
                                              int K, int N) {
  __shared__ bf16 As[2][128][32];   // 2 x 8 KB
  __shared__ bf16 Bs[2][128][32];   // 2 x 8 KB  (row = output col, 32 k-elems)
  const int tid = threadIdx.x, wave = tid >> 5, lane = tid & 31;
  const int lrow = lane & 15, khalf = lane >> 4;
  const int wm = wave >> 2, wn = wave & 3;
  const int bm = blockIdx.y, bn = blockIdx.x;
  const bf16* Abase = A  + (size_t)(bm * 128) * K;
  const bf16* Bbase = BT + (size_t)(bn * 128) * K;

  v8f acc[4][2];
  #pragma unroll
  for (int mi = 0; mi < 4; ++mi)
    #pragma unroll
    for (int nj = 0; nj < 2; ++nj)
      #pragma unroll
      for (int r = 0; r < 8; ++r) acc[mi][nj][r] = 0.0f;

  const int steps = K >> 5;
#if USE_TDM
  if (wave == 0) {
    tdm_load_2d(lds_off32(&As[0][0][0]), Abase, 32, 128, (unsigned long long)K);
    tdm_load_2d(lds_off32(&Bs[0][0][0]), Bbase, 32, 128, (unsigned long long)K);
  }
#endif
  for (int s = 0; s < steps; ++s) {
    const int buf = s & 1;
#if USE_TDM
    if (wave == 0) __builtin_amdgcn_s_wait_tensorcnt(0);
    __syncthreads();                       // tile `buf` visible to all waves
    if (wave == 0 && s + 1 < steps) {      // prefetch next tile into buf^1
      tdm_load_2d(lds_off32(&As[buf ^ 1][0][0]), Abase + (s + 1) * 32,
                  32, 128, (unsigned long long)K);
      tdm_load_2d(lds_off32(&Bs[buf ^ 1][0][0]), Bbase + (s + 1) * 32,
                  32, 128, (unsigned long long)K);
    }
#else
    {
      const int k0 = s * 32;
      for (int i = tid; i < 128 * 32; i += 256) {
        int r = i >> 5, c = i & 31;
        As[buf][r][c] = Abase[(size_t)r * K + k0 + c];
        Bs[buf][r][c] = Bbase[(size_t)r * K + k0 + c];
      }
      __syncthreads();
    }
#endif
    // A frag: lane<16 holds K {0..7,16..23}; lane>=16 holds {8..15,24..31}
    v16bf af[4], bfr[2];
    #pragma unroll
    for (int mi = 0; mi < 4; ++mi) {
      const int row = wm * 64 + mi * 16 + lrow;
      #pragma unroll
      for (int e = 0; e < 16; ++e)
        af[mi][e] = As[buf][row][e + 8 * ((e >> 3) + khalf)];
    }
    // B frag: lane<16 holds K {0..7,16..23}; lane>=16 holds {8..15,24..31}
    #pragma unroll
    for (int nj = 0; nj < 2; ++nj) {
      const int col = wn * 32 + nj * 16 + lrow;
      #pragma unroll
      for (int e = 0; e < 16; ++e)
        bfr[nj][e] = Bs[buf][col][e + 8 * ((e >> 3) + khalf)];
    }
    #pragma unroll
    for (int mi = 0; mi < 4; ++mi)
      #pragma unroll
      for (int nj = 0; nj < 2; ++nj)
        acc[mi][nj] = __builtin_amdgcn_wmma_f32_16x16x32_bf16(
            false, af[mi], false, bfr[nj], (short)0, acc[mi][nj], false, false);
#if !USE_TDM
    __syncthreads();
#endif
  }

  // C layout: VGPR r -> row r + 8*khalf, col = lane&15
  const int rbase = khalf * 8;
  #pragma unroll
  for (int mi = 0; mi < 4; ++mi)
    #pragma unroll
    for (int nj = 0; nj < 2; ++nj)
      #pragma unroll
      for (int r = 0; r < 8; ++r) {
        int row = bm * 128 + wm * 64 + mi * 16 + rbase + r;
        int col = bn * 128 + wn * 32 + nj * 16 + lrow;
        C[(size_t)row * N + col] = acc[mi][nj][r];
      }
}

// ---------------- per-head RMSNorm + RoPE + kv cache scatter ---------------
// grid = T*6, 256 thr (8 waves); wave handles one head-slot (48 total).
// Writes: q bf16 [T,NQ,HD]; k bf16 [T,NKV,HD];
//         v bf16 TRANSPOSED [BATCH,NKV,HD,SEQ]; f32 k/v into kv_buffer rows.
__global__ __launch_bounds__(256) void k_normrope(
    const float* __restrict__ xq, const float* __restrict__ xkv,
    const float* __restrict__ qw, const float* __restrict__ kw,
    const float* __restrict__ cosT, const float* __restrict__ sinT,
    const int* __restrict__ sel,
    bf16* __restrict__ qo, bf16* __restrict__ ko, bf16* __restrict__ voT,
    float* __restrict__ bufo) {
  const int t    = blockIdx.x / 6;
  const int g    = blockIdx.x % 6;
  const int wave = threadIdx.x >> 5;
  const int lane = threadIdx.x & 31;
  const int slot = g * 8 + wave;             // 0..31 q, 32..39 k, 40..47 v
  const int pos  = t & (SEQ - 1);
  const int bb   = t >> 10;                  // batch index
  const float* cr = cosT + (size_t)pos * HD;
  const float* sr = sinT + (size_t)pos * HD;

  if (slot < NQ + NKV) {
    const bool isq = (slot < NQ);
    const int head = isq ? slot : slot - NQ;
    const float* src = isq ? (xq + ((size_t)t * NQ + head) * HD)
                           : (xkv + (size_t)t * (2 * NKV * HD) + head * HD);
    const float* w = isq ? qw : kw;
    float e0 = src[lane], e1 = src[lane + 32], e2 = src[lane + 64], e3 = src[lane + 96];
    float s = e0 * e0 + e1 * e1 + e2 * e2 + e3 * e3;
    #pragma unroll
    for (int o = 16; o > 0; o >>= 1) s += __shfl_xor(s, o);
    const float inv = rsqrtf(s * (1.0f / HD) + 1e-6f);
    float n0 = e0 * inv * w[lane];
    float n1 = e1 * inv * w[lane + 32];
    float n2 = e2 * inv * w[lane + 64];
    float n3 = e3 * inv * w[lane + 96];
    float o0 = n0 * cr[lane]      - n2 * sr[lane];
    float o1 = n1 * cr[lane + 32] - n3 * sr[lane + 32];
    float o2 = n2 * cr[lane + 64] + n0 * sr[lane + 64];
    float o3 = n3 * cr[lane + 96] + n1 * sr[lane + 96];
    if (isq) {
      bf16* dst = qo + ((size_t)t * NQ + head) * HD;
      dst[lane] = (bf16)o0; dst[lane + 32] = (bf16)o1;
      dst[lane + 64] = (bf16)o2; dst[lane + 96] = (bf16)o3;
    } else {
      bf16* dst = ko + ((size_t)t * NKV + head) * HD;
      dst[lane] = (bf16)o0; dst[lane + 32] = (bf16)o1;
      dst[lane + 64] = (bf16)o2; dst[lane + 96] = (bf16)o3;
      float* bd = bufo + ((size_t)sel[t] * (2 * NKV) + head) * HD;
      bd[lane] = o0; bd[lane + 32] = o1; bd[lane + 64] = o2; bd[lane + 96] = o3;
    }
  } else {
    const int head = slot - (NQ + NKV);
    const float* src = xkv + (size_t)t * (2 * NKV * HD) + (NKV * HD) + head * HD;
    float e0 = src[lane], e1 = src[lane + 32], e2 = src[lane + 64], e3 = src[lane + 96];
    const int ss = t & (SEQ - 1);
    bf16* dT = voT + ((size_t)(bb * NKV + head) * HD) * SEQ + ss;   // [d][seq]
    dT[(size_t)(lane)      * SEQ] = (bf16)e0;
    dT[(size_t)(lane + 32) * SEQ] = (bf16)e1;
    dT[(size_t)(lane + 64) * SEQ] = (bf16)e2;
    dT[(size_t)(lane + 96) * SEQ] = (bf16)e3;
    float* bd = bufo + ((size_t)sel[t] * (2 * NKV) + NKV + head) * HD;
    bd[lane] = e0; bd[lane + 32] = e1; bd[lane + 64] = e2; bd[lane + 96] = e3;
  }
}

// ---------------- causal flash attention, GQA (rep=4) ----------------------
// grid = BATCH*NQ*(SEQ/128); block 256 thr. Wave w owns q rows 16w..16w+15.
__global__ __launch_bounds__(256) void k_attn(const bf16* __restrict__ q,
                                              const bf16* __restrict__ k,
                                              const bf16* __restrict__ vT,
                                              bf16* __restrict__ o) {
  __shared__ bf16 Kt[32][128];      // 8 KB  [key][d]
  __shared__ bf16 Vt[128][32];      // 8 KB  [d][key]  (transposed)
  __shared__ float Sc[128][32];     // 16 KB scores
  __shared__ bf16 Pt[128][32];      // 8 KB  probs
  __shared__ float m_s[128], l_s[128], sc_s[128];

  const int qt  = blockIdx.x & 7;
  const int h   = (blockIdx.x >> 3) & 31;
  const int b   = blockIdx.x >> 8;
  const int kvh = h >> 2;
  const int tid = threadIdx.x, wave = tid >> 5, lane = tid & 31;
  const int lrow = lane & 15, khalf = lane >> 4;

  // Q fragments in registers (4 k-steps over head_dim)
  v16bf qf[4];
  {
    const int tokq = b * SEQ + qt * 128 + wave * 16 + lrow;
    const bf16* qp = q + ((size_t)tokq * NQ + h) * HD;
    #pragma unroll
    for (int ks = 0; ks < 4; ++ks)
      #pragma unroll
      for (int e = 0; e < 16; ++e)
        qf[ks][e] = qp[ks * 32 + e + 8 * ((e >> 3) + khalf)];
  }

  v8f oacc[8];
  #pragma unroll
  for (int nt = 0; nt < 8; ++nt)
    #pragma unroll
    for (int r = 0; r < 8; ++r) oacc[nt][r] = 0.0f;

  if (tid < 128) { m_s[tid] = -3.0e38f; l_s[tid] = 0.0f; }
  __syncthreads();

  const int ktmax = (qt * 128 + 127) >> 5;
  for (int kt = 0; kt <= ktmax; ++kt) {
    const int kbase = kt * 32;
#if USE_TDM
    if (wave == 0) {
      tdm_load_2d(lds_off32(&Kt[0][0]),
                  k + ((size_t)(b * SEQ + kbase) * NKV + kvh) * HD,
                  128, 32, (unsigned long long)(NKV * HD));
      tdm_load_2d(lds_off32(&Vt[0][0]),
                  vT + ((size_t)(b * NKV + kvh) * HD) * SEQ + kbase,
                  32, 128, (unsigned long long)SEQ);
      __builtin_amdgcn_s_wait_tensorcnt(0);
    }
#else
    for (int i = tid; i < 32 * 128; i += 256) {
      int r = i >> 7, c = i & 127;
      Kt[r][c] = k[((size_t)(b * SEQ + kbase + r) * NKV + kvh) * HD + c];
    }
    for (int i = tid; i < 128 * 32; i += 256) {
      int r = i >> 5, c = i & 31;
      Vt[r][c] = vT[((size_t)(b * NKV + kvh) * HD + r) * SEQ + kbase + c];
    }
#endif
    __syncthreads();

    // S = Q * K^T (contraction over d, contiguous in Kt rows)
    v8f sacc[2];
    #pragma unroll
    for (int nt = 0; nt < 2; ++nt)
      #pragma unroll
      for (int r = 0; r < 8; ++r) sacc[nt][r] = 0.0f;
    #pragma unroll
    for (int nt = 0; nt < 2; ++nt)
      #pragma unroll
      for (int ks = 0; ks < 4; ++ks) {
        v16bf bfrag;   // B[kk][n] = K[key=nt*16+n][d=ks*32+kk]
        #pragma unroll
        for (int e = 0; e < 16; ++e)
          bfrag[e] = Kt[nt * 16 + lrow][ks * 32 + e + 8 * ((e >> 3) + khalf)];
        sacc[nt] = __builtin_amdgcn_wmma_f32_16x16x32_bf16(
            false, qf[ks], false, bfrag, (short)0, sacc[nt], false, false);
      }
    const float scale = 0.08838834764831845f;  // 1/sqrt(128)
    #pragma unroll
    for (int nt = 0; nt < 2; ++nt)
      #pragma unroll
      for (int r = 0; r < 8; ++r)
        Sc[wave * 16 + khalf * 8 + r][nt * 16 + lrow] = sacc[nt][r] * scale;
    __syncthreads();

    // online softmax (one thread per q row)
    if (tid < 128) {
      const int row = tid;
      const int vlim = (qt * 128 + row) - kbase;   // cols c <= vlim valid
      float m = m_s[row], l = l_s[row];
      float rmax = -3.0e38f;
      #pragma unroll
      for (int c = 0; c < 32; ++c)
        if (c <= vlim) rmax = fmaxf(rmax, Sc[row][c]);
      const float mnew = fmaxf(m, rmax);
      const float scl  = __expf(m - mnew);
      float psum = 0.0f;
      #pragma unroll
      for (int c = 0; c < 32; ++c) {
        float p = (c <= vlim) ? __expf(Sc[row][c] - mnew) : 0.0f;
        Pt[row][c] = (bf16)p;
        psum += p;
      }
      m_s[row]  = mnew;
      l_s[row]  = l * scl + psum;
      sc_s[row] = scl;
    }
    __syncthreads();

    // O = O*scl + P * V  (V frag contiguous thanks to transposed Vt)
    v16bf pf;
    #pragma unroll
    for (int e = 0; e < 16; ++e)
      pf[e] = Pt[wave * 16 + lrow][e + 8 * ((e >> 3) + khalf)];
    #pragma unroll
    for (int nt = 0; nt < 8; ++nt) {
      #pragma unroll
      for (int r = 0; r < 8; ++r)
        oacc[nt][r] *= sc_s[wave * 16 + khalf * 8 + r];
      v16bf vfrag;   // B[kk][n] = V[key=kk][d=nt*16+n] = Vt[d][key]
      #pragma unroll
      for (int e = 0; e < 16; ++e)
        vfrag[e] = Vt[nt * 16 + lrow][e + 8 * ((e >> 3) + khalf)];
      oacc[nt] = __builtin_amdgcn_wmma_f32_16x16x32_bf16(
          false, pf, false, vfrag, (short)0, oacc[nt], false, false);
    }
    __syncthreads();
  }

  // normalize, store bf16 attention output [T, NQ*HD]
  #pragma unroll
  for (int nt = 0; nt < 8; ++nt)
    #pragma unroll
    for (int r = 0; r < 8; ++r) {
      int row = wave * 16 + khalf * 8 + r;
      int tok = b * SEQ + qt * 128 + row;
      float val = oacc[nt][r] / l_s[row];
      o[(size_t)tok * HID + h * HD + nt * 16 + lrow] = (bf16)val;
    }
}

// ---------------------------------------------------------------------------
extern "C" void kernel_launch(void* const* d_in, const int* in_sizes, int n_in,
                              void* d_out, int out_size, void* d_ws, size_t ws_size,
                              hipStream_t stream) {
  (void)in_sizes; (void)n_in; (void)out_size; (void)ws_size;
  const float* x    = (const float*)d_in[0];
  const float* qw   = (const float*)d_in[1];
  const float* kvw  = (const float*)d_in[2];
  const float* ow   = (const float*)d_in[3];
  const float* qnw  = (const float*)d_in[4];
  const float* knw  = (const float*)d_in[5];
  const float* cosT = (const float*)d_in[6];
  const float* sinT = (const float*)d_in[7];
  const float* kvb  = (const float*)d_in[8];
  const int*   sel  = (const int*)d_in[9];

  float* out  = (float*)d_out;                 // [T, HID]
  float* bufo = out + (size_t)T * HID;         // [KVROWS, 2*NKV, HD]

  // workspace carve (~184 MB)
  char* w = (char*)d_ws;
  bf16*  xb    = (bf16*)w;  w += (size_t)T * HID * 2;
  bf16*  qwT   = (bf16*)w;  w += (size_t)HID * HID * 2;            // [N,K]
  bf16*  kvwT  = (bf16*)w;  w += (size_t)HID * (2 * NKV * HD) * 2; // [N,K]
  bf16*  owT   = (bf16*)w;  w += (size_t)HID * HID * 2;            // [N,K]
  float* xqf   = (float*)w; w += (size_t)T * NQ * HD * 4;
  float* xkvf  = (float*)w; w += (size_t)T * 2 * NKV * HD * 4;
  bf16*  qb    = (bf16*)w;  w += (size_t)T * NQ * HD * 2;
  bf16*  kb    = (bf16*)w;  w += (size_t)T * NKV * HD * 2;
  bf16*  vbT   = (bf16*)w;  w += (size_t)T * NKV * HD * 2;         // [B,NKV,HD,SEQ]
  bf16*  ab    = (bf16*)w;  w += (size_t)T * HID * 2;

  // 1) kv_buffer pass-through
  (void)hipMemcpyAsync(bufo, kvb, (size_t)KVROWS * 2 * NKV * HD * sizeof(float),
                       hipMemcpyDeviceToDevice, stream);

  // 2) bf16 conversions (weights transposed to [N,K] for the GEMM B-side)
  {
    int n = T * HID;
    k_cvt<<<dim3((n + 255) / 256), 256, 0, stream>>>(x, xb, n);
  }
  k_cvt_t<<<dim3((HID * HID + 255) / 256), 256, 0, stream>>>(qw, qwT, HID, HID);
  k_cvt_t<<<dim3((HID * (2 * NKV * HD) + 255) / 256), 256, 0, stream>>>(
      kvw, kvwT, HID, 2 * NKV * HD);
  k_cvt_t<<<dim3((HID * HID + 255) / 256), 256, 0, stream>>>(ow, owT, HID, HID);

  // 3) projections (TDM-staged WMMA GEMM)
  k_gemm<<<dim3(HID / 128, T / 128), 256, 0, stream>>>(xb, qwT, xqf, HID, HID);
  k_gemm<<<dim3((2 * NKV * HD) / 128, T / 128), 256, 0, stream>>>(
      xb, kvwT, xkvf, HID, 2 * NKV * HD);

  // 4) q/k RMSNorm + RoPE + kv-cache scatter (+ V transposed for attention)
  k_normrope<<<dim3(T * 6), 256, 0, stream>>>(xqf, xkvf, qnw, knw, cosT, sinT,
                                              sel, qb, kb, vbT, bufo);

  // 5) causal flash attention (TDM-staged, WMMA)
  k_attn<<<dim3(BATCH * NQ * (SEQ / 128)), 256, 0, stream>>>(qb, kb, vbT, ab);

  // 6) output projection straight into d_out
  k_gemm<<<dim3(HID / 128, T / 128), 256, 0, stream>>>(ab, owT, out, HID, HID);
}
